// CausalSelfAttention_66254165508788
// MI455X (gfx1250) — matrix operand order repaired and checked by reading, not compiled, not run on hardware
//
#include <hip/hip_runtime.h>

typedef unsigned short u16;
typedef __attribute__((ext_vector_type(16))) __bf16 v16bf;
typedef __attribute__((ext_vector_type(8)))  float  v8f;
typedef __attribute__((ext_vector_type(4)))  unsigned int u32x4;

union Frag {
  v16bf bf;
  u32x4 q[2];
  u16   h[16];
};

__device__ __forceinline__ u16 f2bf(float f) {
  union { float f; unsigned u; } v; v.f = f;
  unsigned r = v.u + 0x7FFFu + ((v.u >> 16) & 1u);  // round-to-nearest-even
  return (u16)(r >> 16);
}

// LDS byte offset of a __shared__ object: flat-LDS aperture keeps the
// workgroup-relative offset in addr[31:0].
__device__ __forceinline__ unsigned ldsoff(const void* p) {
  return (unsigned)(unsigned long long)p;
}

// CDNA5 async global->LDS DMA, 16B per lane, tracked by ASYNCcnt (no VGPR transit).
__device__ __forceinline__ void async_ld16(unsigned ldsDst, const void* g) {
  asm volatile("global_load_async_to_lds_b128 %0, %1, off"
               :: "v"(ldsDst), "v"((unsigned long long)g) : "memory");
}
__device__ __forceinline__ void wait_async0() {
  asm volatile("s_wait_asynccnt 0x0" ::: "memory");
}

// Problem constants
constexpr int BATCH = 2;
constexpr int SEQ   = 2048;
constexpr int NH    = 32;
constexpr int NKV   = 8;
constexpr int HDIM  = 64;
constexpr int EMB   = 2048;
constexpr int NQKV  = (NH + 2 * NKV) * HDIM;  // 3072
constexpr int MTOT  = BATCH * SEQ;            // 4096

// ---------------------------------------------------------------------------
// fp32 -> bf16 elementwise convert (grid-stride)
// ---------------------------------------------------------------------------
__global__ void cvt_bf16(const float* __restrict__ in, u16* __restrict__ out, long n) {
  long i = blockIdx.x * (long)blockDim.x + threadIdx.x;
  long stride = gridDim.x * (long)blockDim.x;
  for (; i < n; i += stride) out[i] = f2bf(in[i]);
}

// ---------------------------------------------------------------------------
// C[M,N] (f32) = A[M,K] (bf16, row-major) * B[N,K]^T (bf16, row-major)
// 4 waves / block.  Block tile: 64(M) x 128(N), k-step 32.
// A(64x32) and B(128x32) staged in double-buffered LDS (padded rows) via
// CDNA5 GLOBAL_LOAD_ASYNC_TO_LDS_B128 (ASYNCcnt), DMA'd during the WMMAs.
// A sched_barrier pins all 18 fragment ds_loads (distinct registers) ahead of
// the 8 WMMAs so dscnt waits pipeline instead of waiting to zero per WMMA.
// Each wave owns a 16(M) x 128(N) strip.
// ---------------------------------------------------------------------------
constexpr int LDP = 40;  // padded LDS row stride in u16 (32 data + 8 pad)
constexpr unsigned ABUF = 64 * LDP * 2;    // bytes per A buffer (5120)
constexpr unsigned BBUF = 128 * LDP * 2;   // bytes per B buffer (10240)

__global__ __launch_bounds__(128) void gemm_bf16_nt(const u16* __restrict__ A,
                                                    const u16* __restrict__ B,
                                                    float* __restrict__ C,
                                                    int M, int N, int K) {
  __shared__ u16 lA[2][64 * LDP];    // ~10 KB
  __shared__ u16 lB[2][128 * LDP];   // ~20 KB
  const int tid  = threadIdx.x;
  const int wave = tid >> 5;
  const int lane = tid & 31;
  const int m0 = blockIdx.y * 64;
  const int n0 = blockIdx.x * 128;
  const int r    = lane & 15;
  const int hi   = lane >> 4;
  const int kbA  = hi * 16;   // A fragment: one contiguous 16-elem chunk
  const int kbB  = hi * 16;   // B fragment: one contiguous 16-elem chunk
  const int rowb = hi * 8;    // C fragment row base

  // cooperative tile-load mapping: 16B chunks, 4 chunks per 32-elem row
  const int lrow = tid >> 2;        // 0..31
  const int lko  = (tid & 3) * 8;   // 0,8,16,24

  const u16* Ag = A + (size_t)(m0 + lrow) * K + lko;
  const u16* Bg = B + (size_t)(n0 + lrow) * K + lko;
  const size_t rstep = (size_t)32 * K;   // 32 rows

  // per-thread LDS destinations (buffer 0); buffer 1 = +ABUF/+BBUF
  const unsigned la0 = ldsoff(&lA[0][lrow * LDP + lko]);
  const unsigned lb0 = ldsoff(&lB[0][lrow * LDP + lko]);
  const unsigned rowq = 32 * LDP * 2;  // 32 LDS rows in bytes

  v8f acc[8] = {};

  // prologue: async-stage k=0 tile into buffer 0
  async_ld16(la0,            Ag);
  async_ld16(la0 + rowq,     Ag + rstep);
  async_ld16(lb0,            Bg);
  async_ld16(lb0 + rowq,     Bg + rstep);
  async_ld16(lb0 + 2 * rowq, Bg + 2 * rstep);
  async_ld16(lb0 + 3 * rowq, Bg + 3 * rstep);
  wait_async0();
  __syncthreads();

  int buf = 0;
  for (int k = 0; k < K; k += 32) {
    const bool more = (k + 32) < K;
    if (more) {  // DMA next tile into the other buffer; overlaps the WMMAs below
      const unsigned nb = (unsigned)(buf ^ 1);
      const unsigned da = la0 + nb * ABUF;
      const unsigned db = lb0 + nb * BBUF;
      async_ld16(da,            Ag + k + 32);
      async_ld16(da + rowq,     Ag + rstep + k + 32);
      async_ld16(db,            Bg + k + 32);
      async_ld16(db + rowq,     Bg + rstep + k + 32);
      async_ld16(db + 2 * rowq, Bg + 2 * rstep + k + 32);
      async_ld16(db + 3 * rowq, Bg + 3 * rstep + k + 32);
    }

    // issue ALL fragment ds_loads first (distinct registers), then the WMMAs.
    const u16* la = &lA[buf][(wave * 16 + r) * LDP];
    Frag a, bm[8];
    a.q[0] = *(const u32x4*)(la + kbA);
    a.q[1] = *(const u32x4*)(la + kbA + 8);
#pragma unroll
    for (int j = 0; j < 8; ++j) {
      const u16* lb = &lB[buf][(j * 16 + r) * LDP + kbB];
      bm[j].q[0] = *(const u32x4*)(lb);
      bm[j].q[1] = *(const u32x4*)(lb + 8);
    }
    // forbid the scheduler from sinking the ds_loads back into the WMMA
    // sequence (which would coalesce registers and force wait-to-zero).
    __builtin_amdgcn_sched_barrier(0);
#pragma unroll
    for (int j = 0; j < 8; ++j)
      acc[j] = __builtin_amdgcn_wmma_f32_16x16x32_bf16(
          false, a.bf, false, bm[j].bf, (short)0, acc[j], false, false);

    if (more) {
      wait_async0();     // our 6 DMAs for the next buffer are done
      __syncthreads();   // everyone finished reading + everyone's DMAs done
      buf ^= 1;
    }
  }

  const int mw = m0 + wave * 16;
#pragma unroll
  for (int j = 0; j < 8; ++j)
#pragma unroll
    for (int i = 0; i < 8; ++i)
      C[(size_t)(mw + rowb + i) * N + n0 + j * 16 + r] = acc[j][i];
}

// ---------------------------------------------------------------------------
// RoPE on Q and K heads of the fp32 QKV buffer; writes bf16 Q [B,NH,T,HD]
// and bf16 K [B,NKV,T,HD]. One thread per (b,t,head,pair).
// ---------------------------------------------------------------------------
__global__ void rope_qk(const float* __restrict__ qkv,
                        const float* __restrict__ fc, const float* __restrict__ fs,
                        u16* __restrict__ Qb, u16* __restrict__ Kb) {
  long idx = blockIdx.x * (long)blockDim.x + threadIdx.x;
  const long total = (long)BATCH * SEQ * (NH + NKV) * (HDIM / 2);
  if (idx >= total) return;
  int j = (int)(idx & 31); long t1 = idx >> 5;
  int hh = (int)(t1 % (NH + NKV)); t1 /= (NH + NKV);
  int t = (int)(t1 % SEQ);
  int b = (int)(t1 / SEQ);
  const float* row = qkv + ((size_t)b * SEQ + t) * NQKV + hh * HDIM;
  float x0 = row[2 * j], x1 = row[2 * j + 1];
  float c = fc[t * (HDIM / 2) + j], s = fs[t * (HDIM / 2) + j];
  float o0 = x0 * c - x1 * s;
  float o1 = x0 * s + x1 * c;
  u16* dst;
  if (hh < NH) dst = Qb + (((size_t)b * NH + hh) * SEQ + t) * HDIM;
  else         dst = Kb + (((size_t)b * NKV + (hh - NH)) * SEQ + t) * HDIM;
  dst[2 * j]     = f2bf(o0);
  dst[2 * j + 1] = f2bf(o1);
}

// ---------------------------------------------------------------------------
// V copy/convert, transposed to [B, NKV, HD, T] so that attention's V
// B-fragments (k = seq index) load contiguously.
// ---------------------------------------------------------------------------
__global__ void v_transpose(const float* __restrict__ qkv, u16* __restrict__ Vt) {
  long idx = blockIdx.x * (long)blockDim.x + threadIdx.x;
  const long total = (long)BATCH * SEQ * NKV * HDIM;
  if (idx >= total) return;
  int d = (int)(idx & 63); long t1 = idx >> 6;
  int kh = (int)(t1 % NKV); t1 /= NKV;
  int t = (int)(t1 % SEQ);
  int b = (int)(t1 / SEQ);
  float v = qkv[((size_t)b * SEQ + t) * NQKV + (NH + NKV + kh) * HDIM + d];
  Vt[(((size_t)b * NKV + kh) * HDIM + d) * SEQ + t] = f2bf(v);
}

// ---------------------------------------------------------------------------
// Flash attention: one wave per 16-row Q tile (full HD=64), s-chunks of 32.
// K fragments are preloaded (sched_barrier-pinned) before the 4 score WMMAs;
// V fragment loads are pinned above the softmax so ~100 VALU/trans ops hide
// their global latency.  Online softmax via width-16 xor-shuffles; P
// transposed C->A layout through per-wave LDS.  GQA: kv head = h/4.
// Output bf16 [B,T,EMB].
// ---------------------------------------------------------------------------
__global__ __launch_bounds__(128) void attn_kernel(const u16* __restrict__ Qb,
                                                   const u16* __restrict__ Kb,
                                                   const u16* __restrict__ Vt,
                                                   u16* __restrict__ Yb) {
  __shared__ u16 plds[4][16 * 32];
  const int wave = threadIdx.x >> 5;
  const int lane = threadIdx.x & 31;
  const int tile = blockIdx.x * 4 + wave;
  const int tilesPerBH = SEQ / 16;           // 128
  const int bh = tile / tilesPerBH;
  const int tt = tile - bh * tilesPerBH;
  const int b  = bh >> 5;
  const int h  = bh & 31;
  const int kh = h >> 2;
  const int t0 = tt * 16;
  const int r    = lane & 15;
  const int hi   = lane >> 4;
  const int kbA  = hi * 16;
  const int kbB  = hi * 16;
  const int rowb = hi * 8;

  const u16* Q = Qb + ((size_t)(b * NH + h) * SEQ) * HDIM;
  const u16* K = Kb + ((size_t)(b * NKV + kh) * SEQ) * HDIM;
  const u16* V = Vt + ((size_t)(b * NKV + kh) * HDIM) * SEQ;

  // Q A-fragments for d in [0,32) and [32,64)
  Frag qa[2];
  const u16* qrow = Q + (size_t)(t0 + r) * HDIM;
  qa[0].q[0] = *(const u32x4*)(qrow + kbA);
  qa[0].q[1] = *(const u32x4*)(qrow + kbA + 8);
  qa[1].q[0] = *(const u32x4*)(qrow + 32 + kbA);
  qa[1].q[1] = *(const u32x4*)(qrow + 32 + kbA + 8);

  v8f o[4] = {};
  float mrow[8], lrow[8];
#pragma unroll
  for (int i = 0; i < 8; ++i) { mrow[i] = -1e30f; lrow[i] = 0.f; }
  const float sc = 0.125f;  // 1/sqrt(64)
  const int smax = t0 + 15;
  u16* pl = plds[wave];

  for (int sb = 0; sb <= smax; sb += 32) {
    // prefetch next s-chunk of K and V (global_prefetch_b8)
    if (sb + 32 <= smax) {
      __builtin_prefetch(K + (size_t)(sb + 32 + r) * HDIM, 0, 0);
      __builtin_prefetch(V + (size_t)r * SEQ + sb + 32, 0, 0);
    }

    // ---- preload all 4 K fragments (pinned), then 4 score WMMAs ----
    Frag kf[2][2];  // [d-half][col-tile]
#pragma unroll
    for (int half = 0; half < 2; ++half) {
      const u16* k0 = K + (size_t)(sb + r) * HDIM + half * 32;
      const u16* k1 = K + (size_t)(sb + 16 + r) * HDIM + half * 32;
      kf[half][0].q[0] = *(const u32x4*)(k0 + kbB);
      kf[half][0].q[1] = *(const u32x4*)(k0 + kbB + 8);
      kf[half][1].q[0] = *(const u32x4*)(k1 + kbB);
      kf[half][1].q[1] = *(const u32x4*)(k1 + kbB + 8);
    }
    __builtin_amdgcn_sched_barrier(0);
    v8f s0 = {}, s1 = {};
#pragma unroll
    for (int half = 0; half < 2; ++half) {
      s0 = __builtin_amdgcn_wmma_f32_16x16x32_bf16(
          false, qa[half].bf, false, kf[half][0].bf, (short)0, s0, false, false);
      s1 = __builtin_amdgcn_wmma_f32_16x16x32_bf16(
          false, qa[half].bf, false, kf[half][1].bf, (short)0, s1, false, false);
    }

    // ---- issue V fragment loads now (pinned above the softmax) ----
    Frag vf[4];
#pragma unroll
    for (int f = 0; f < 4; ++f) {
      const u16* vp = V + (size_t)(f * 16 + r) * SEQ + sb;
      vf[f].q[0] = *(const u32x4*)(vp + kbB);
      vf[f].q[1] = *(const u32x4*)(vp + kbB + 8);
    }
    __builtin_amdgcn_sched_barrier(0);

    // ---- causal mask + online softmax (row groups = 16-lane halves) ----
#pragma unroll
    for (int i = 0; i < 8; ++i) {
      const int trow = t0 + rowb + i;
      const int c0 = sb + r;
      const int c1 = sb + 16 + r;
      float v0 = (c0 <= trow) ? s0[i] * sc : -1e30f;
      float v1 = (c1 <= trow) ? s1[i] * sc : -1e30f;
      float cm = fmaxf(v0, v1);
      cm = fmaxf(cm, __shfl_xor(cm, 1, 16));
      cm = fmaxf(cm, __shfl_xor(cm, 2, 16));
      cm = fmaxf(cm, __shfl_xor(cm, 4, 16));
      cm = fmaxf(cm, __shfl_xor(cm, 8, 16));
      float mn   = fmaxf(mrow[i], cm);
      float corr = __expf(mrow[i] - mn);
      mrow[i] = mn;
      float p0 = __expf(v0 - mn);
      float p1 = __expf(v1 - mn);
      float ps = p0 + p1;
      ps += __shfl_xor(ps, 1, 16);
      ps += __shfl_xor(ps, 2, 16);
      ps += __shfl_xor(ps, 4, 16);
      ps += __shfl_xor(ps, 8, 16);
      lrow[i] = lrow[i] * corr + ps;
      o[0][i] *= corr; o[1][i] *= corr; o[2][i] *= corr; o[3][i] *= corr;
      // P (C-fragment layout) -> LDS, row-major 16x32 bf16
      pl[(rowb + i) * 32 + r]      = f2bf(p0);
      pl[(rowb + i) * 32 + 16 + r] = f2bf(p1);
    }

    __builtin_amdgcn_wave_barrier();  // keep LDS store->load order (in-order per wave)
    Frag pf;  // P as A-fragment 16x32
    pf.q[0] = *(const u32x4*)(pl + r * 32 + kbA);
    pf.q[1] = *(const u32x4*)(pl + r * 32 + kbA + 8);
    __builtin_amdgcn_wave_barrier();

    // ---- O(16x64) += P(16x32) @ V(32x64) with preloaded V fragments ----
#pragma unroll
    for (int f = 0; f < 4; ++f)
      o[f] = __builtin_amdgcn_wmma_f32_16x16x32_bf16(
          false, pf.bf, false, vf[f].bf, (short)0, o[f], false, false);
  }

  // ---- normalize and store to [B,T,EMB] bf16, heads interleaved ----
#pragma unroll
  for (int i = 0; i < 8; ++i) {
    const float inv = 1.f / lrow[i];
    const int trow = t0 + rowb + i;
    u16* yrow = Yb + ((size_t)b * SEQ + trow) * EMB + h * HDIM;
#pragma unroll
    for (int f = 0; f < 4; ++f)
      yrow[f * 16 + r] = f2bf(o[f][i] * inv);
  }
}

// ---------------------------------------------------------------------------
extern "C" void kernel_launch(void* const* d_in, const int* in_sizes, int n_in,
                              void* d_out, int out_size, void* d_ws, size_t ws_size,
                              hipStream_t stream) {
  const float* x     = (const float*)d_in[0];
  const float* wqkv  = (const float*)d_in[1];
  const float* wproj = (const float*)d_in[2];
  const float* fc    = (const float*)d_in[3];
  const float* fs    = (const float*)d_in[4];
  // d_in[5] = start_pos (unused: full-sequence prefill)

  char* p = (char*)d_ws;
  u16*   x_bf     = (u16*)p;   p += (size_t)MTOT * EMB * 2;
  u16*   wqkv_bf  = (u16*)p;   p += (size_t)NQKV * EMB * 2;
  u16*   wproj_bf = (u16*)p;   p += (size_t)EMB * EMB * 2;
  float* qkv      = (float*)p; p += (size_t)MTOT * NQKV * 4;
  u16*   Qb       = (u16*)p;   p += (size_t)BATCH * NH * SEQ * HDIM * 2;
  u16*   Kb       = (u16*)p;   p += (size_t)BATCH * NKV * SEQ * HDIM * 2;
  u16*   Vt       = (u16*)p;   p += (size_t)BATCH * NKV * HDIM * SEQ * 2;
  u16*   Yb       = (u16*)p;   p += (size_t)MTOT * EMB * 2;

  cvt_bf16<<<2048, 256, 0, stream>>>(x, x_bf, (long)MTOT * EMB);
  cvt_bf16<<<2048, 256, 0, stream>>>(wqkv, wqkv_bf, (long)NQKV * EMB);
  cvt_bf16<<<2048, 256, 0, stream>>>(wproj, wproj_bf, (long)EMB * EMB);

  // QKV = x @ Wqkv^T
  gemm_bf16_nt<<<dim3(NQKV / 128, MTOT / 64), 128, 0, stream>>>(
      x_bf, wqkv_bf, qkv, MTOT, NQKV, EMB);

  {
    long n = (long)BATCH * SEQ * (NH + NKV) * (HDIM / 2);
    rope_qk<<<(unsigned)((n + 255) / 256), 256, 0, stream>>>(qkv, fc, fs, Qb, Kb);
  }
  {
    long n = (long)BATCH * SEQ * NKV * HDIM;
    v_transpose<<<(unsigned)((n + 255) / 256), 256, 0, stream>>>(qkv, Vt);
  }

  attn_kernel<<<(BATCH * NH * (SEQ / 16)) / 4, 128, 0, stream>>>(Qb, Kb, Vt, Yb);

  // out = y @ Wproj^T  (fp32 output)
  gemm_bf16_nt<<<dim3(EMB / 128, MTOT / 64), 128, 0, stream>>>(
      Yb, wproj_bf, (float*)d_out, MTOT, EMB, EMB);
}
